// MultiHeadAttention_35527969472545
// MI455X (gfx1250) — compile-verified, matrix operand order checked by reading, not run on hardware
//
#include <hip/hip_runtime.h>
#include <hip/hip_bf16.h>

// ---------------------------------------------------------------------------
// MHA forward for B=2, S=2048, D=1024, H=16, DK=64 on gfx1250 (wave32, WMMA).
// Pipeline: cvt q/k/v + weights -> bf16 | 3x projection GEMM (pure-bf16 WMMA)
// scattering to head-major layouts | flash-attention (WMMA bf16, online
// softmax, LDS P relayout) | output projection GEMM (WMMA bf16 -> f32 out).
// ---------------------------------------------------------------------------

#define SQ   2048
#define DM   1024
#define NH   16
#define DK   64
#define NB   2
#define MROWS (NB * SQ)   // 4096

typedef __attribute__((ext_vector_type(16))) __bf16 bf16x16;
typedef __attribute__((ext_vector_type(8)))  float  f32x8;

union FragAB {
  bf16x16 v;
  uint4 q[2];
  unsigned short us[16];
};

__device__ __forceinline__ unsigned short f2bf_bits(float f) {
  unsigned u = __builtin_bit_cast(unsigned, f);
  unsigned r = u + 0x7FFFu + ((u >> 16) & 1u);  // round-to-nearest-even
  return (unsigned short)(r >> 16);
}

__device__ __forceinline__ f32x8 wmma_bf16(const FragAB& a, const FragAB& b, f32x8 c) {
  return __builtin_amdgcn_wmma_f32_16x16x32_bf16(
      /*neg_a=*/false, a.v, /*neg_b=*/false, b.v,
      /*c_mod=*/(short)0, c, /*reuse_a=*/false, /*reuse_b=*/false);
}

// --------------------------- fp32 -> bf16 convert ---------------------------
__global__ void __launch_bounds__(256)
cvt_kernel(const float* __restrict__ src, unsigned short* __restrict__ dst, int n4) {
  int i = blockIdx.x * blockDim.x + threadIdx.x;
  if (i < n4) {
    float4 f = reinterpret_cast<const float4*>(src)[i];
    ushort4 o;
    o.x = f2bf_bits(f.x); o.y = f2bf_bits(f.y);
    o.z = f2bf_bits(f.z); o.w = f2bf_bits(f.w);
    reinterpret_cast<ushort4*>(dst)[i] = o;
  }
}

// ------------- projection GEMM: Y = X @ W^T (pure-bf16 WMMA) ---------------
// mode 0: dst = Qh [b,h,s,dk] * (1/sqrt(dk));  mode 1: dst = Kh [b,h,s,dk]
// mode 2: dst = Vt [b,h,dk,s]
// One wave computes a 16x64 output tile. 4096 waves total (256 m-tiles x 16 n-tiles).
__global__ void __launch_bounds__(256)
proj_qkv_kernel(const unsigned short* __restrict__ Xb, const unsigned short* __restrict__ Wb,
                unsigned short* __restrict__ dst, int mode) {
  const int lane = threadIdx.x & 31;
  const int wave = threadIdx.x >> 5;
  const int wid  = blockIdx.x * 8 + wave;
  const int mt = wid >> 4;      // 0..255
  const int nt = wid & 15;      // 0..15
  const int m0 = mt * 16;
  const int n0 = nt * 64;
  const int half = lane >> 4;
  const int lr   = lane & 15;

  f32x8 acc[4] = {};
  const unsigned short* arow = Xb + (size_t)(m0 + lr) * DM;

  for (int k = 0; k < DM; k += 32) {
    FragAB a;
    a.q[0] = *(const uint4*)(arow + k + half * 8);
    a.q[1] = *(const uint4*)(arow + k + 16 + half * 8);
#pragma unroll
    for (int f = 0; f < 4; ++f) {
      FragAB bf;
      const unsigned short* brow = Wb + (size_t)(n0 + f * 16 + lr) * DM + k;
      bf.q[0] = *(const uint4*)(brow + half * 8);
      bf.q[1] = *(const uint4*)(brow + 16 + half * 8);
      acc[f] = wmma_bf16(a, bf, acc[f]);
    }
  }

  const float scale = (mode == 0) ? 0.125f : 1.0f;  // 1/sqrt(64)
#pragma unroll
  for (int f = 0; f < 4; ++f) {
#pragma unroll
    for (int r = 0; r < 8; ++r) {
      int m = m0 + r + half * 8;
      int b = m >> 11;
      int s = m & (SQ - 1);
      int n = n0 + f * 16 + lr;
      int h = n & 15;       // head index is fast axis of the (dk,h) split
      int i = n >> 4;       // dk index
      size_t idx;
      if (mode == 2) idx = ((size_t)((b * NH + h) * DK + i)) * SQ + s;
      else           idx = ((size_t)((b * NH + h) * SQ + s)) * DK + i;
      dst[idx] = f2bf_bits(acc[f][r] * scale);
    }
  }
}

// --------------------------- flash attention -------------------------------
// One wave per (b, h, 16-row q-tile). Online softmax over 32-key chunks.
__global__ void __launch_bounds__(256)
attn_kernel(const unsigned short* __restrict__ Qh, const unsigned short* __restrict__ Kh,
            const unsigned short* __restrict__ Vt, unsigned short* __restrict__ Ob) {
  __shared__ unsigned short Plds[8][16 * 32];  // per-wave 16x32 bf16 P tile

  const int lane = threadIdx.x & 31;
  const int wave = threadIdx.x >> 5;
  const int wid  = blockIdx.x * 8 + wave;      // 0..4095
  const int qt = wid & 127;
  const int h  = (wid >> 7) & 15;
  const int b  = wid >> 11;
  const int q0 = qt * 16;
  const int half = lane >> 4;
  const int lr   = lane & 15;

  const unsigned short* Qbase = Qh + ((size_t)(b * NH + h) * SQ) * DK;
  const unsigned short* Kbase = Kh + ((size_t)(b * NH + h) * SQ) * DK;
  const unsigned short* Vbase = Vt + ((size_t)(b * NH + h) * DK) * SQ;
  unsigned short* P = &Plds[wave][0];

  // Q fragments for this tile: rows q0..q0+15, two k-chunks (dk 0..31, 32..63)
  FragAB qa[2];
  {
    const unsigned short* qrow = Qbase + (size_t)(q0 + lr) * DK;
#pragma unroll
    for (int kk = 0; kk < 2; ++kk) {
      qa[kk].q[0] = *(const uint4*)(qrow + kk * 32 + half * 8);
      qa[kk].q[1] = *(const uint4*)(qrow + kk * 32 + 16 + half * 8);
    }
  }

  f32x8 o[4] = {};
  float mst[8], lst[8];
#pragma unroll
  for (int r = 0; r < 8; ++r) { mst[r] = -INFINITY; lst[r] = 0.0f; }

  for (int kt = 0; kt < q0 + 16; kt += 32) {
    __builtin_prefetch((const void*)(Kbase + (size_t)(kt + 32 + lr) * DK), 0, 1);

    // logits S = Q * K^T : two 16x16 fragments (keys kt..+15, kt+16..+31)
    f32x8 sc[2] = {};
#pragma unroll
    for (int f = 0; f < 2; ++f) {
      const unsigned short* kr = Kbase + (size_t)(kt + f * 16 + lr) * DK;
#pragma unroll
      for (int kk = 0; kk < 2; ++kk) {
        FragAB bf;
        bf.q[0] = *(const uint4*)(kr + kk * 32 + half * 8);
        bf.q[1] = *(const uint4*)(kr + kk * 32 + 16 + half * 8);
        sc[f] = wmma_bf16(qa[kk], bf, sc[f]);
      }
    }

    // causal mask (only diagonal chunks need it)
    if (kt + 32 > q0) {
#pragma unroll
      for (int f = 0; f < 2; ++f)
#pragma unroll
        for (int r = 0; r < 8; ++r) {
          int key = kt + f * 16 + lr;
          int qq  = q0 + r + half * 8;
          if (key > qq) sc[f][r] = -1e30f;
        }
    }

    // online softmax update + stage P into LDS (C-layout -> row-major bf16)
#pragma unroll
    for (int r = 0; r < 8; ++r) {
      float cm = fmaxf(sc[0][r], sc[1][r]);
      cm = fmaxf(cm, __shfl_xor(cm, 8, 32));
      cm = fmaxf(cm, __shfl_xor(cm, 4, 32));
      cm = fmaxf(cm, __shfl_xor(cm, 2, 32));
      cm = fmaxf(cm, __shfl_xor(cm, 1, 32));
      float nm   = fmaxf(mst[r], cm);
      float corr = __expf(mst[r] - nm);
      float p0 = __expf(sc[0][r] - nm);
      float p1 = __expf(sc[1][r] - nm);
      float rs = p0 + p1;
      rs += __shfl_xor(rs, 8, 32);
      rs += __shfl_xor(rs, 4, 32);
      rs += __shfl_xor(rs, 2, 32);
      rs += __shfl_xor(rs, 1, 32);
      lst[r] = lst[r] * corr + rs;
      mst[r] = nm;
#pragma unroll
      for (int f = 0; f < 4; ++f) o[f][r] *= corr;
      int row = r + half * 8;
      P[row * 32 + lr]      = f2bf_bits(p0);
      P[row * 32 + 16 + lr] = f2bf_bits(p1);
    }
    asm volatile("s_wait_dscnt 0" ::: "memory");

    // reload P in A-matrix layout (16x32, k = key positions)
    FragAB pa;
    {
      const unsigned short* prow = P + lr * 32;
      pa.q[0] = *(const uint4*)(prow + half * 8);
      pa.q[1] = *(const uint4*)(prow + 16 + half * 8);
    }

    // O += P * V  (V columns are key-contiguous thanks to Vt layout)
#pragma unroll
    for (int f = 0; f < 4; ++f) {
      FragAB vb;
      const unsigned short* vcol = Vbase + (size_t)(f * 16 + lr) * SQ + kt;
      vb.q[0] = *(const uint4*)(vcol + half * 8);
      vb.q[1] = *(const uint4*)(vcol + 16 + half * 8);
      o[f] = wmma_bf16(pa, vb, o[f]);
    }
  }

  // normalize and scatter to Ob [b, s, dk*H + h] (bf16)
#pragma unroll
  for (int r = 0; r < 8; ++r) {
    float inv = 1.0f / lst[r];
    int s = q0 + r + half * 8;
#pragma unroll
    for (int f = 0; f < 4; ++f) {
      int dk = f * 16 + lr;
      int n  = dk * NH + h;
      Ob[((size_t)(b * SQ + s)) * DM + n] = f2bf_bits(o[f][r] * inv);
    }
  }
}

// ---------------------- output projection: out = Ob @ Wo^T ------------------
__global__ void __launch_bounds__(256)
oproj_kernel(const unsigned short* __restrict__ Ob, const unsigned short* __restrict__ Wob,
             float* __restrict__ out) {
  const int lane = threadIdx.x & 31;
  const int wave = threadIdx.x >> 5;
  const int wid  = blockIdx.x * 8 + wave;
  const int mt = wid >> 4;
  const int nt = wid & 15;
  const int m0 = mt * 16;
  const int n0 = nt * 64;
  const int half = lane >> 4;
  const int lr   = lane & 15;

  f32x8 acc[4] = {};
  const unsigned short* arow = Ob + (size_t)(m0 + lr) * DM;

  for (int k = 0; k < DM; k += 32) {
    FragAB a;
    a.q[0] = *(const uint4*)(arow + k + half * 8);
    a.q[1] = *(const uint4*)(arow + k + 16 + half * 8);
#pragma unroll
    for (int f = 0; f < 4; ++f) {
      FragAB bf;
      const unsigned short* brow = Wob + (size_t)(n0 + f * 16 + lr) * DM + k;
      bf.q[0] = *(const uint4*)(brow + half * 8);
      bf.q[1] = *(const uint4*)(brow + 16 + half * 8);
      acc[f] = wmma_bf16(a, bf, acc[f]);
    }
  }

#pragma unroll
  for (int f = 0; f < 4; ++f)
#pragma unroll
    for (int r = 0; r < 8; ++r) {
      int m = m0 + r + half * 8;
      int n = n0 + f * 16 + lr;
      out[(size_t)m * DM + n] = acc[f][r];
    }
}

// ---------------------------------------------------------------------------
extern "C" void kernel_launch(void* const* d_in, const int* in_sizes, int n_in,
                              void* d_out, int out_size, void* d_ws, size_t ws_size,
                              hipStream_t stream) {
  const float* q  = (const float*)d_in[0];
  const float* k  = (const float*)d_in[1];
  const float* v  = (const float*)d_in[2];
  // d_in[3] is the causal tril mask; applied analytically in attn_kernel.
  const float* wq = (const float*)d_in[4];
  const float* wk = (const float*)d_in[5];
  const float* wv = (const float*)d_in[6];
  const float* wo = (const float*)d_in[7];
  float* out = (float*)d_out;

  // Scratch layout (bf16 elements). Ob aliases qb: qb is dead once the Q
  // projection has completed, and attn_kernel (which writes Ob) is ordered
  // after it on the stream.
  unsigned short* ws  = (unsigned short*)d_ws;
  unsigned short* wqb = ws;                     // 1M bf16 each
  unsigned short* wkb = wqb + (1u << 20);
  unsigned short* wvb = wkb + (1u << 20);
  unsigned short* wob = wvb + (1u << 20);
  unsigned short* qb  = wob + (1u << 20);       // 4M bf16 each
  unsigned short* kb  = qb  + (4u << 20);
  unsigned short* vb  = kb  + (4u << 20);
  unsigned short* Qh  = vb  + (4u << 20);
  unsigned short* Kh  = Qh  + (4u << 20);
  unsigned short* Vt  = Kh  + (4u << 20);
  unsigned short* Ob  = qb;                     // alias (qb dead after proj Q)
  // total footprint: (4 + 6*4) M bf16 = 56 MB

  const int n4w = (DM * DM) / 4;                // 262144 float4's per weight
  const int n4x = (MROWS * DM) / 4;             // 1048576 float4's per activation
  dim3 cblk(256);
  cvt_kernel<<<dim3((n4w + 255) / 256), cblk, 0, stream>>>(wq, wqb, n4w);
  cvt_kernel<<<dim3((n4w + 255) / 256), cblk, 0, stream>>>(wk, wkb, n4w);
  cvt_kernel<<<dim3((n4w + 255) / 256), cblk, 0, stream>>>(wv, wvb, n4w);
  cvt_kernel<<<dim3((n4w + 255) / 256), cblk, 0, stream>>>(wo, wob, n4w);
  cvt_kernel<<<dim3((n4x + 255) / 256), cblk, 0, stream>>>(q, qb, n4x);
  cvt_kernel<<<dim3((n4x + 255) / 256), cblk, 0, stream>>>(k, kb, n4x);
  cvt_kernel<<<dim3((n4x + 255) / 256), cblk, 0, stream>>>(v, vb, n4x);

  dim3 pblk(256), pgrid(512);                   // 4096 waves = 256x16 tiles
  proj_qkv_kernel<<<pgrid, pblk, 0, stream>>>(qb, wqb, Qh, 0);
  proj_qkv_kernel<<<pgrid, pblk, 0, stream>>>(kb, wkb, Kh, 1);
  proj_qkv_kernel<<<pgrid, pblk, 0, stream>>>(vb, wvb, Vt, 2);

  attn_kernel<<<dim3(512), dim3(256), 0, stream>>>(Qh, Kh, Vt, Ob);

  oproj_kernel<<<pgrid, pblk, 0, stream>>>(Ob, wob, out);
}
